// SparseMultiHeadAttention_3582002725048
// MI455X (gfx1250) — matrix operand order repaired and checked by reading, not compiled, not run on hardware
//
#include <hip/hip_runtime.h>
#include <hip/hip_bf16.h>
#include <math.h>

// ---------------- problem constants ----------------
#define BN    2
#define SEQ   2048
#define EMB   1024
#define NH    16
#define DH    64
#define ROWS  (BN * SEQ)        // 4096 token rows
#define HALF  16                // WIN//2
#define WINSZ 33

typedef __attribute__((ext_vector_type(16))) __bf16 v16bf;
typedef __attribute__((ext_vector_type(8)))  __bf16 v8bf;
typedef __attribute__((ext_vector_type(8)))  float  v8f;

static __device__ __forceinline__ v16bf concat8(v8bf lo, v8bf hi) {
  return __builtin_shufflevector(lo, hi, 0, 1, 2, 3, 4, 5, 6, 7,
                                         8, 9, 10, 11, 12, 13, 14, 15);
}

// =====================================================================
// f32 -> bf16 row-major conversion (8 elems / thread, b128 in, b128 out)
// =====================================================================
__global__ void cvt_to_bf16(const float* __restrict__ s,
                            __bf16* __restrict__ d, long n8)
{
  long i = (long)blockIdx.x * blockDim.x + threadIdx.x;
  const long stride = (long)gridDim.x * blockDim.x;
  for (; i < n8; i += stride) {
    const float4* s4 = (const float4*)s + i * 2;
    const float4 x0 = s4[0];
    const float4 x1 = s4[1];
    v8bf o;
    o[0] = (__bf16)x0.x; o[1] = (__bf16)x0.y; o[2] = (__bf16)x0.z; o[3] = (__bf16)x0.w;
    o[4] = (__bf16)x1.x; o[5] = (__bf16)x1.y; o[6] = (__bf16)x1.z; o[7] = (__bf16)x1.w;
    ((v8bf*)d)[i] = o;
  }
}

// =====================================================================
// GEMM: Y[4096,1024] = X[4096,1024] @ W[1024,1024]^T + bias
// (bf16 in, WMMA f32 acc). One wave computes 16(M) x 64(N).
// Software-pipelined, copy-free double buffering. The k-loop is kept
// rolled (#pragma unroll 1) so f0/f1 stay live across the back-edge:
// loads for the next half-step issue a full 4-WMMA group ahead,
// giving partial s_wait_loadcnt instead of wait-0 per WMMA.
// A frag: lane = row, lanes 0-15 hold K[0..15] contiguous,
// lanes 16-31 K[16..31]. B frag: lane = column, lanes 0-15
// hold K[0..15] contiguous, lanes 16-31 K[16..31].
// =====================================================================
struct FragSet {
  v8bf a_lo, a_hi;
  v8bf b_lo[4], b_hi[4];
};

static __device__ __forceinline__ void load_frags(
    FragSet& f, const __bf16* xrow,
    const __bf16* wr0, const __bf16* wr1,
    const __bf16* wr2, const __bf16* wr3, int k)
{
  f.a_lo    = *(const v8bf*)(xrow + k);
  f.a_hi    = *(const v8bf*)(xrow + k + 8);
  f.b_lo[0] = *(const v8bf*)(wr0 + k);
  f.b_hi[0] = *(const v8bf*)(wr0 + k + 8);
  f.b_lo[1] = *(const v8bf*)(wr1 + k);
  f.b_hi[1] = *(const v8bf*)(wr1 + k + 8);
  f.b_lo[2] = *(const v8bf*)(wr2 + k);
  f.b_hi[2] = *(const v8bf*)(wr2 + k + 8);
  f.b_lo[3] = *(const v8bf*)(wr3 + k);
  f.b_hi[3] = *(const v8bf*)(wr3 + k + 8);
}

static __device__ __forceinline__ void do_wmma(v8f acc[4], const FragSet& f)
{
  const v16bf a = concat8(f.a_lo, f.a_hi);
#pragma unroll
  for (int t = 0; t < 4; ++t) {
    const v16bf b = concat8(f.b_lo[t], f.b_hi[t]);
    acc[t] = __builtin_amdgcn_wmma_f32_16x16x32_bf16(
        /*neg_a=*/false, a, /*neg_b=*/false, b,
        /*c_mod=*/(short)0, acc[t], /*reuse_a=*/false, /*reuse_b=*/false);
  }
}

__global__ void gemm_bf16_wmma(const __bf16* __restrict__ X,
                               const __bf16* __restrict__ W,
                               const float* __restrict__ bias,
                               float* __restrict__ Y)
{
  const int K = EMB, N = EMB;              // compile-time dims
  const int lane = threadIdx.x & 31;
  const int wid  = threadIdx.x >> 5;
  const int mt   = blockIdx.x * 8 + wid;   // 16-row tile index
  const int nb   = blockIdx.y * 64;        // 64-col tile base
  const int l15  = lane & 15;
  const int hi   = lane >> 4;              // lane half select

  const __bf16* xrow = X + (size_t)(mt * 16 + l15) * K + hi * 16;
  const __bf16* wr0 = W + (size_t)(nb +  0 + l15) * K + hi * 16;
  const __bf16* wr1 = W + (size_t)(nb + 16 + l15) * K + hi * 16;
  const __bf16* wr2 = W + (size_t)(nb + 32 + l15) * K + hi * 16;
  const __bf16* wr3 = W + (size_t)(nb + 48 + l15) * K + hi * 16;

  v8f acc[4] = {};

  FragSet f0, f1;
  load_frags(f0, xrow, wr0, wr1, wr2, wr3, 0);
  load_frags(f1, xrow, wr0, wr1, wr2, wr3, 32);

  // steady state: no guards needed (k0 <= K-128 => k0+96 <= K-32 in range)
#pragma unroll 1
  for (int k0 = 0; k0 < K - 64; k0 += 64) {
    do_wmma(acc, f0);                                    // consume f0 (k0)
    load_frags(f0, xrow, wr0, wr1, wr2, wr3, k0 + 64);   // refill, overlaps wmma(f1)
    do_wmma(acc, f1);                                    // consume f1 (k0+32)
    load_frags(f1, xrow, wr0, wr1, wr2, wr3, k0 + 96);   // overlaps next wmma(f0)
    if (k0 + 128 < K) {
      __builtin_prefetch(xrow + k0 + 128, 0, 1);         // global_prefetch_b8
      __builtin_prefetch(wr0 + k0 + 128, 0, 1);
    }
  }
  // epilogue: last 64 of K
  do_wmma(acc, f0);
  do_wmma(acc, f1);

  // C/D layout: VGPR v -> M = v + 8*hi, N = lane&15
#pragma unroll
  for (int t = 0; t < 4; ++t) {
    const int col = nb + t * 16 + l15;
    const float bv = bias[col];
#pragma unroll
    for (int v = 0; v < 8; ++v) {
      const int r = mt * 16 + v + hi * 8;
      Y[(size_t)r * N + col] = acc[t][v] + bv;
    }
  }
}

// =====================================================================
// Zero-fill (float4 grid-stride) for the dense attn output (512 MiB)
// =====================================================================
__global__ void zero_f4(float4* __restrict__ p, long n4)
{
  long i = (long)blockIdx.x * blockDim.x + threadIdx.x;
  const long stride = (long)gridDim.x * blockDim.x;
  const float4 z = make_float4(0.f, 0.f, 0.f, 0.f);
  for (; i < n4; i += stride) p[i] = z;
}

// =====================================================================
// Banded attention: one thread per (b,h,q). 33-wide window, fully
// unrolled so arrays stay in VGPRs. Loads batched before FMA chains.
// =====================================================================
__global__ void banded_attn(const float* __restrict__ Qp,
                            const float* __restrict__ Kp,
                            const float* __restrict__ Vp,
                            float* __restrict__ attn,
                            float* __restrict__ ctx)
{
  const int idx = blockIdx.x * blockDim.x + threadIdx.x; // (b*NH+h)*SEQ + q
  const int q   = idx & (SEQ - 1);
  const int bh  = idx >> 11;
  const int h   = bh & (NH - 1);
  const int b   = bh >> 4;
  const size_t headbase = (size_t)(b * SEQ) * EMB + (size_t)h * DH;

  const float4* q4 = (const float4*)(Qp + headbase + (size_t)q * EMB);
  float4 qv[16];
#pragma unroll
  for (int i = 0; i < 16; ++i) qv[i] = q4[i];

  float e[WINSZ];
  float smax = -INFINITY;
#pragma unroll
  for (int o = 0; o < WINSZ; ++o) {
    const int j = q - HALF + o;
    const bool valid = (unsigned)j < (unsigned)SEQ;
    const int js = valid ? j : q;  // safe address for masked offsets
    const float4* k4 = (const float4*)(Kp + headbase + (size_t)js * EMB);
    float4 kk[16];
#pragma unroll
    for (int i = 0; i < 16; ++i) kk[i] = k4[i];   // batch loads first
    float s = 0.f;
#pragma unroll
    for (int i = 0; i < 16; ++i)
      s += qv[i].x * kk[i].x + qv[i].y * kk[i].y +
           qv[i].z * kk[i].z + qv[i].w * kk[i].w;
    s *= 0.125f;                   // 1/sqrt(Dh=64)
    e[o] = valid ? s : -INFINITY;
    smax = fmaxf(smax, e[o]);
  }

  float denom = 0.f;
#pragma unroll
  for (int o = 0; o < WINSZ; ++o) { e[o] = __expf(e[o] - smax); denom += e[o]; }
  const float inv = 1.f / denom;

  float* arow = attn + (size_t)idx * SEQ;
  float4 outv[16];
#pragma unroll
  for (int i = 0; i < 16; ++i) outv[i] = make_float4(0.f, 0.f, 0.f, 0.f);

#pragma unroll
  for (int o = 0; o < WINSZ; ++o) {
    const int j = q - HALF + o;
    if ((unsigned)j < (unsigned)SEQ) {
      const float w = e[o] * inv;
      arow[j] = w;
      const float4* v4 = (const float4*)(Vp + headbase + (size_t)j * EMB);
      float4 vv[16];
#pragma unroll
      for (int i = 0; i < 16; ++i) vv[i] = v4[i];
#pragma unroll
      for (int i = 0; i < 16; ++i) {
        outv[i].x += w * vv[i].x; outv[i].y += w * vv[i].y;
        outv[i].z += w * vv[i].z; outv[i].w += w * vv[i].w;
      }
    }
  }

  float4* c4 = (float4*)(ctx + headbase + (size_t)q * EMB);
#pragma unroll
  for (int i = 0; i < 16; ++i) c4[i] = outv[i];
}

// =====================================================================
// Residual + LayerNorm over E=1024. One 256-thread block (8 waves) per
// row; wave32 shuffle reduce then LDS combine.
// =====================================================================
__global__ void resid_layernorm(const float* __restrict__ Qin,
                                const float* __restrict__ O,
                                const float* __restrict__ gamma,
                                const float* __restrict__ beta,
                                float* __restrict__ Y)
{
  __shared__ float ssum[8];
  __shared__ float ssq[8];
  const int row = blockIdx.x;
  const int tid = threadIdx.x;
  const size_t base = (size_t)row * EMB + (size_t)tid * 4;

  float x[4];
  float s = 0.f, s2 = 0.f;
#pragma unroll
  for (int i = 0; i < 4; ++i) {
    x[i] = Qin[base + i] + O[base + i];
    s += x[i]; s2 += x[i] * x[i];
  }
#pragma unroll
  for (int m = 16; m >= 1; m >>= 1) {
    s  += __shfl_xor(s,  m, 32);
    s2 += __shfl_xor(s2, m, 32);
  }
  if ((tid & 31) == 0) { ssum[tid >> 5] = s; ssq[tid >> 5] = s2; }
  __syncthreads();
  float ts = 0.f, ts2 = 0.f;
#pragma unroll
  for (int w = 0; w < 8; ++w) { ts += ssum[w]; ts2 += ssq[w]; }

  const float mu  = ts  * (1.f / EMB);
  const float var = ts2 * (1.f / EMB) - mu * mu;
  const float rs  = rsqrtf(var + 1e-5f);
#pragma unroll
  for (int i = 0; i < 4; ++i)
    Y[base + i] = (x[i] - mu) * rs * gamma[tid * 4 + i] + beta[tid * 4 + i];
}

// =====================================================================
extern "C" void kernel_launch(void* const* d_in, const int* in_sizes, int n_in,
                              void* d_out, int out_size, void* d_ws, size_t ws_size,
                              hipStream_t stream)
{
  const float* Q_in  = (const float*)d_in[0];
  const float* K_in  = (const float*)d_in[1];
  const float* V_in  = (const float*)d_in[2];
  const float* Wq    = (const float*)d_in[3];
  const float* bq    = (const float*)d_in[4];
  const float* Wk    = (const float*)d_in[5];
  const float* bk    = (const float*)d_in[6];
  const float* Wv    = (const float*)d_in[7];
  const float* bv    = (const float*)d_in[8];
  const float* Wo    = (const float*)d_in[9];
  const float* bo    = (const float*)d_in[10];
  const float* gamma = (const float*)d_in[11];
  const float* beta  = (const float*)d_in[12];

  float* y    = (float*)d_out;                  // [B,N,E]
  float* attn = y + (size_t)ROWS * EMB;         // [B,H,N,N]

  // workspace layout
  float* ws = (float*)d_ws;
  const size_t mat  = (size_t)ROWS * EMB;       // 4 Mi floats
  const size_t wmat = (size_t)EMB * EMB;        // 1 Mi floats
  float* Qp  = ws;
  float* Kp  = ws + 1 * mat;
  float* Vp  = ws + 2 * mat;
  float* ctx = ws + 3 * mat;
  float* Op  = ws + 4 * mat;
  __bf16* bfb    = (__bf16*)(ws + 5 * mat);
  __bf16* Qin_bf = bfb;
  __bf16* Kin_bf = bfb + 1 * mat;
  __bf16* Vin_bf = bfb + 2 * mat;
  __bf16* ctx_bf = bfb + 3 * mat;
  __bf16* Wq_bf  = bfb + 4 * mat;
  __bf16* Wk_bf  = Wq_bf + 1 * wmat;
  __bf16* Wv_bf  = Wq_bf + 2 * wmat;
  __bf16* Wo_bf  = Wq_bf + 3 * wmat;

  // ---- pre-convert activations & weights to bf16 ----
  cvt_to_bf16<<<2048, 256, 0, stream>>>(Q_in, Qin_bf, (long)(mat / 8));
  cvt_to_bf16<<<2048, 256, 0, stream>>>(K_in, Kin_bf, (long)(mat / 8));
  cvt_to_bf16<<<2048, 256, 0, stream>>>(V_in, Vin_bf, (long)(mat / 8));
  cvt_to_bf16<<<512,  256, 0, stream>>>(Wq, Wq_bf, (long)(wmat / 8));
  cvt_to_bf16<<<512,  256, 0, stream>>>(Wk, Wk_bf, (long)(wmat / 8));
  cvt_to_bf16<<<512,  256, 0, stream>>>(Wv, Wv_bf, (long)(wmat / 8));
  cvt_to_bf16<<<512,  256, 0, stream>>>(Wo, Wo_bf, (long)(wmat / 8));

  const dim3 gb(256);
  const dim3 gg(ROWS / 128, EMB / 64);          // 32 x 16 blocks

  // ---- Q/K/V projections (x @ W^T + b), bf16 WMMA ----
  gemm_bf16_wmma<<<gg, gb, 0, stream>>>(Qin_bf, Wq_bf, bq, Qp);
  gemm_bf16_wmma<<<gg, gb, 0, stream>>>(Kin_bf, Wk_bf, bk, Kp);
  gemm_bf16_wmma<<<gg, gb, 0, stream>>>(Vin_bf, Wv_bf, bv, Vp);

  // ---- dense attn: zero everything, write band weights ----
  const long attn_n4 = (long)BN * NH * SEQ * SEQ / 4;
  zero_f4<<<4096, 256, 0, stream>>>((float4*)attn, attn_n4);
  banded_attn<<<(BN * NH * SEQ) / 256, 256, 0, stream>>>(Qp, Kp, Vp, attn, ctx);

  // ---- output projection + residual/LayerNorm ----
  cvt_to_bf16<<<2048, 256, 0, stream>>>(ctx, ctx_bf, (long)(mat / 8));
  gemm_bf16_wmma<<<gg, gb, 0, stream>>>(ctx_bf, Wo_bf, bo, Op);
  resid_layernorm<<<ROWS, 256, 0, stream>>>(Q_in, Op, gamma, beta, y);
}
